// WaveNet_87265145520880
// MI455X (gfx1250) — matrix operand order repaired and checked by reading, not compiled, not run on hardware
//
#include <hip/hip_runtime.h>

#define B_N    4
#define T_MELC 63
#define M_CH   80
#define HOP_C  256
#define K_UPC  800
#define R_CH   120
#define S_CH   240
#define Q_CH   256
#define NB_C   16
#define L_LEN  15872      // (T_MEL-1)*HOP
#define PADUP  400        // (K_UP + 4*HOP - WIN)/2

typedef __attribute__((ext_vector_type(16))) __bf16          v16bf;
typedef __attribute__((ext_vector_type(8)))  float           v8f;
typedef __attribute__((ext_vector_type(4)))  float           f32x4;
typedef __attribute__((ext_vector_type(8)))  unsigned short  u16x8;

union BF16x16 { v16bf v; unsigned short u[16]; };

__device__ __forceinline__ unsigned short f2bf(float f) {
  union { float f; unsigned int u; } x; x.f = f;
  return (unsigned short)((x.u + 0x7FFFu + ((x.u >> 16) & 1u)) >> 16);  // RNE
}

__device__ __forceinline__ u16x8 zero_u16x8() {
  u16x8 v;
  #pragma unroll
  for (int j = 0; j < 8; ++j) v[j] = 0;
  return v;
}

// Raw LDS byte offset of a __shared__ object (addrspace(3) ptrtoint)
__device__ __forceinline__ unsigned lds_off(const void* p) {
  return (unsigned)(size_t)(const __attribute__((address_space(3))) void*)p;
}

// Async 16B global->LDS copy (GLOBAL_LOAD_ASYNC_TO_LDS_B128, ASYNCcnt-tracked)
__device__ __forceinline__ void async_copy_b128(unsigned lds_byte, const void* gsrc) {
  asm volatile("global_load_async_to_lds_b128 %0, %1, off"
               :: "v"(lds_byte), "v"(gsrc) : "memory");
}

__device__ __forceinline__ void wait_asynccnt0() {
  asm volatile("s_wait_asynccnt 0x0" ::: "memory");
}

// A-matrix (16x32 bf16) per-lane K mapping, contiguous: k = 16*hi + e (matches B)
__device__ __forceinline__ int kmapA(int e, int hi) {
  return e + (hi << 4);
}

// ---------------------------------------------------------------------------
// Pack weight matrices into WMMA A-fragment layout (bf16), one tile per block.
// kind 0: fused gate(+cond) A: M=240, K=352 = [res(t) | res(t-d) | cond96]
// kind 1: skip A: M=240,K=128   kind 2: res A: M=128,K=128
// kind 3: h1 A: M=256,K=256     kind 4: h2 A: M=256,K=256
// ---------------------------------------------------------------------------
__global__ void pack_frags(int kind, int mt_n, int kt_n,
                           const float* __restrict__ w_gate,
                           const float* __restrict__ w_cond,
                           const float* __restrict__ w_skip,
                           const float* __restrict__ w_res,
                           const float* __restrict__ w_h1,
                           const float* __restrict__ w_h2,
                           unsigned short* __restrict__ dst)
{
  int tile = blockIdx.x;
  int kt = tile % kt_n;
  int mt = (tile / kt_n) % mt_n;
  int i  = tile / (kt_n * mt_n);
  int tid  = threadIdx.x;            // 512 threads: (lane, elem)
  int lane = tid >> 4, e = tid & 15;
  int hi = lane >> 4, laneLo = lane & 15;
  int m = mt * 16 + laneLo;
  int k = kt * 32 + kmapA(e, hi);
  float w = 0.f;
  if (kind == 0) {
    if (k < 128)      { if (k < R_CH)        w = w_gate[(((size_t)i*S_CH + m)*R_CH + k)*2 + 1]; }
    else if (k < 256) { int r = k - 128; if (r < R_CH) w = w_gate[(((size_t)i*S_CH + m)*R_CH + r)*2 + 0]; }
    else              { int c = k - 256; if (c < M_CH) w = w_cond[((size_t)i*S_CH + m)*M_CH + c]; }
  } else if (kind == 1) {
    if (k < R_CH) w = w_skip[((size_t)i*S_CH + m)*R_CH + k];
  } else if (kind == 2) {
    if (m < R_CH && k < R_CH) w = w_res[((size_t)i*R_CH + m)*R_CH + k];
  } else if (kind == 3) {
    if (k < S_CH) w = w_h1[(size_t)m*S_CH + k];
  } else {
    w = w_h2[(size_t)m*Q_CH + k];
  }
  dst[(size_t)tile*512 + lane*16 + e] = f2bf(w);
}

// Combined gate+cond bias
__global__ void pack_bias(const float* __restrict__ bg, const float* __restrict__ bc,
                          float* __restrict__ dst)
{
  int i = blockIdx.x * 256 + threadIdx.x;
  if (i < NB_C * S_CH) dst[i] = bg[i] + bc[i];
}

// ---------------------------------------------------------------------------
// Upsampler: cond[b,t,co] = b_up[co] + sum_{j,ci} mel[b,ci,j]*w_up[ci,co,t+400-256j]
// Stored bf16, channel-contiguous [b][t][96] with pad channels 80..95 zero.
// ---------------------------------------------------------------------------
__global__ void upsample_kernel(const float* __restrict__ mel,
                                const float* __restrict__ w_up,
                                const float* __restrict__ b_up,
                                unsigned short* __restrict__ cond)
{
  int b = blockIdx.y;
  long long flat = (long long)blockIdx.x * blockDim.x + threadIdx.x;
  if (flat >= (long long)96 * L_LEN) return;
  int co = (int)(flat % 96);
  int t  = (int)(flat / 96);
  float acc = 0.f;
  if (co < M_CH) {
    const float* melb = mel + (size_t)b * M_CH * T_MELC;
    int jlo = (t > PADUP - 1) ? ((t - (PADUP - 1) + HOP_C - 1) >> 8) : 0;
    int jhi = (t + PADUP) >> 8; if (jhi > T_MELC - 1) jhi = T_MELC - 1;
    acc = b_up[co];
    for (int j = jlo; j <= jhi; ++j) {
      int tap = t + PADUP - j * HOP_C;
      const float* wp = w_up + (size_t)co * K_UPC + tap;
      for (int ci = 0; ci < M_CH; ++ci)
        acc += melb[ci * T_MELC + j] * wp[(size_t)ci * M_CH * K_UPC];
    }
  }
  cond[((size_t)b * L_LEN + t) * 96 + co] = f2bf(acc);
}

// ---------------------------------------------------------------------------
// Init: res = w_in*wav + b_in, written to BOTH representations:
//   bf16 shadow  [b][t][128]         (pad ch 120..127 = 0)   -> WMMA B staging
//   f32 C-tiled  [b][mt(8)][t][16]   (pad rows = 0)          -> WMMA C operand
// ---------------------------------------------------------------------------
__global__ void init_res_kernel(const float* __restrict__ wav,
                                const float* __restrict__ w_in,
                                const float* __restrict__ b_in,
                                unsigned short* __restrict__ res_bf,
                                float* __restrict__ res_t)
{
  int b = blockIdx.y;
  long long flat = (long long)blockIdx.x * blockDim.x + threadIdx.x;
  if (flat >= (long long)L_LEN * 128) return;
  int c = (int)(flat & 127);
  int t = (int)(flat >> 7);
  float v = (c < R_CH) ? (w_in[c] * wav[(size_t)b * L_LEN + t] + b_in[c]) : 0.f;
  res_bf[((size_t)b * L_LEN + t) * 128 + c] = f2bf(v);
  res_t[(((size_t)b * 8 + (c >> 4)) * L_LEN + t) * 16 + (c & 15)] = v;
}

// ---------------------------------------------------------------------------
// One WaveNet block for 16 time steps per workgroup (8 waves).
// ---------------------------------------------------------------------------
__global__ __launch_bounds__(256) void wavenet_block_kernel(
    const unsigned short* __restrict__ res_bf_in,   // [b][t][128] bf16
    unsigned short* __restrict__ res_bf_out,
    const float* __restrict__ res_t_in,             // [b][8][t][16] f32
    float* __restrict__ res_t_out,
    float* __restrict__ skip_t,                     // [b][15][t][16] f32
    const unsigned short* __restrict__ cond,        // [b][t][96] bf16
    const unsigned short* __restrict__ gateF,
    const unsigned short* __restrict__ skipF,
    const unsigned short* __restrict__ resF,
    const float* __restrict__ bgc,                  // combined gate+cond bias
    const float* __restrict__ bs, const float* __restrict__ br,
    int dil, int first)
{
  __shared__ unsigned short act[16][368];   // [n][K-row], K=352 padded
  __shared__ float gated[S_CH][16];
  __shared__ unsigned short zsh[16][144];   // [n][z-row], 128 padded

  int b  = blockIdx.y;
  int t0 = blockIdx.x * 16;
  int tid = threadIdx.x;
  int lane = tid & 31, wave = tid >> 5;
  int hi = lane >> 4, laneLo = lane & 15;

  const unsigned short* rbf = res_bf_in + (size_t)b * L_LEN * 128;
  const unsigned short* cbf = cond + (size_t)b * L_LEN * 96;

  // Stage B-operand via async global->LDS copies (ASYNCcnt path):
  // act[n][0..127]=res(t), [128..255]=res(t-d), [256..351]=cond.
  // Channel-contiguous layouts -> pure 16B copies, data never touches VGPRs.
  for (int idx = tid; idx < 16 * 44; idx += 256) {
    int n = idx / 44, ch = idx % 44;     // ch = 16-byte chunk (8 x u16)
    int t = t0 + n;
    unsigned dstb = lds_off(&act[n][ch * 8]);
    if (ch < 16) {
      async_copy_b128(dstb, rbf + ((size_t)t << 7) + ch * 8);
    } else if (ch < 32) {
      int td = t - dil;
      if (td >= 0) async_copy_b128(dstb, rbf + ((size_t)td << 7) + (ch - 16) * 8);
      else         *(u16x8*)&act[n][ch * 8] = zero_u16x8();
    } else {
      async_copy_b128(dstb, cbf + (size_t)t * 96 + (ch - 32) * 8);
    }
  }
  wait_asynccnt0();
  __syncthreads();

  // B fragments: lane = time column; lanes 0-15 K=0..15, 16-31 K=16..31 (contig)
  BF16x16 bfr[11];
  #pragma unroll
  for (int kt = 0; kt < 11; ++kt)
    bfr[kt].v = *(const v16bf*)&act[laneLo][kt * 32 + 16 * hi];

  // Gated GEMM: 15 M-tiles over 8 waves (wave-uniform guard keeps EXEC all-1)
  #pragma unroll
  for (int which = 0; which < 2; ++which) {
    int mt = wave + which * 8;
    if (mt < 15) {
      v8f acc;
      #pragma unroll
      for (int r = 0; r < 8; ++r) acc[r] = 0.f;
      const unsigned short* af = gateF + (size_t)mt * 11 * 512 + lane * 16;
      #pragma unroll
      for (int kt = 0; kt < 11; ++kt) {
        BF16x16 a; a.v = *(const v16bf*)(af + kt * 512);
        acc = __builtin_amdgcn_wmma_f32_16x16x32_bf16(
                  false, a.v, false, bfr[kt].v, (short)0, acc, false, false);
      }
      #pragma unroll
      for (int r = 0; r < 8; ++r) {
        int c = mt * 16 + r + hi * 8;
        gated[c][laneLo] = acc[r] + bgc[c];
      }
    }
  }
  __syncthreads();

  // z = tanh(f) * sigmoid(g), stored bf16 with zero pad rows 120..127
  for (int idx = tid; idx < 128 * 16; idx += 256) {
    int row = idx >> 4, n = idx & 15;
    unsigned short h = 0;
    if (row < R_CH) {
      float f = gated[row][n];
      float g = gated[row + R_CH][n];
      float z = tanhf(f) * (1.f / (1.f + __expf(-g)));
      h = f2bf(z);
    }
    zsh[n][row] = h;
  }
  __syncthreads();

  BF16x16 zb[4];
  #pragma unroll
  for (int kt = 0; kt < 4; ++kt)
    zb[kt].v = *(const v16bf*)&zsh[laneLo][kt * 32 + 16 * hi];

  // skip += Ws*z + bs   (C/D live in WMMA-tiled layout: 32B/lane load+store)
  #pragma unroll
  for (int which = 0; which < 2; ++which) {
    int mt = wave + which * 8;
    if (mt < 15) {
      float* skp = skip_t + ((((size_t)b * 15 + mt) * L_LEN + t0 + laneLo) * 16) + 8 * hi;
      v8f acc;
      if (first) {
        #pragma unroll
        for (int r = 0; r < 8; ++r) acc[r] = 0.f;
      } else {
        acc = *(const v8f*)skp;
      }
      const unsigned short* af = skipF + (size_t)mt * 4 * 512 + lane * 16;
      #pragma unroll
      for (int kt = 0; kt < 4; ++kt) {
        BF16x16 a; a.v = *(const v16bf*)(af + kt * 512);
        acc = __builtin_amdgcn_wmma_f32_16x16x32_bf16(
                  false, a.v, false, zb[kt].v, (short)0, acc, false, false);
      }
      v8f sv;
      #pragma unroll
      for (int r = 0; r < 8; ++r) sv[r] = acc[r] + bs[mt * 16 + r + hi * 8];
      *(v8f*)skp = sv;
    }
  }

  // res_out = res_in + Wr*z + br  (8 M-tiles, one per wave; tile 7 half-pad)
  {
    int mt = wave;
    const float* rip = res_t_in  + ((((size_t)b * 8 + mt) * L_LEN + t0 + laneLo) * 16) + 8 * hi;
    float*       rop = res_t_out + ((((size_t)b * 8 + mt) * L_LEN + t0 + laneLo) * 16) + 8 * hi;
    v8f acc = *(const v8f*)rip;     // pad rows contain zeros (never stored anyway)
    const unsigned short* af = resF + (size_t)mt * 4 * 512 + lane * 16;
    #pragma unroll
    for (int kt = 0; kt < 4; ++kt) {
      BF16x16 a; a.v = *(const v16bf*)(af + kt * 512);
      acc = __builtin_amdgcn_wmma_f32_16x16x32_bf16(
                false, a.v, false, zb[kt].v, (short)0, acc, false, false);
    }
    v8f sv; u16x8 bv;
    #pragma unroll
    for (int r = 0; r < 8; ++r) {
      int c = mt * 16 + r + hi * 8;
      float v = (c < R_CH) ? (acc[r] + br[c]) : 0.f;
      sv[r] = v;
      bv[r] = f2bf(v);
    }
    *(v8f*)rop = sv;
    *(u16x8*)(res_bf_out + ((size_t)b * L_LEN + t0 + laneLo) * 128 + mt * 16 + 8 * hi) = bv;
  }
}

// ---------------------------------------------------------------------------
// Head: logits = W2*relu(W1*relu(skip)+b1)+b2, fully fused per 16-step tile.
// ---------------------------------------------------------------------------
__global__ __launch_bounds__(256) void head_kernel(
    const float* __restrict__ skip_t,               // [b][15][t][16] f32
    const unsigned short* __restrict__ h1F,
    const unsigned short* __restrict__ h2F,
    const float* __restrict__ b_h1, const float* __restrict__ b_h2,
    float* __restrict__ out)
{
  __shared__ unsigned short act[16][256];   // relu(skip), K=240 padded to 256
  __shared__ unsigned short hsh[16][272];   // relu(h), 256 channels
  __shared__ float lsh[16][264];            // logits staging for coalesced store

  int b  = blockIdx.y;
  int t0 = blockIdx.x * 16;
  int tid = threadIdx.x;
  int lane = tid & 31, wave = tid >> 5;
  int hi = lane >> 4, laneLo = lane & 15;

  // Stage relu(skip) from the C-tiled layout (16B f32x4 loads)
  for (int idx = tid; idx < 15 * 16 * 4; idx += 256) {
    int q = idx & 3, n = (idx >> 2) & 15, mt = idx >> 6;
    f32x4 s = *(const f32x4*)(skip_t + (((size_t)b * 15 + mt) * L_LEN + t0 + n) * 16 + q * 4);
    #pragma unroll
    for (int j = 0; j < 4; ++j) {
      float v = s[j] > 0.f ? s[j] : 0.f;
      act[n][mt * 16 + q * 4 + j] = f2bf(v);
    }
  }
  for (int idx = tid; idx < 16 * 16; idx += 256)   // zero pad channels 240..255
    act[idx >> 4][240 + (idx & 15)] = 0;
  __syncthreads();

  BF16x16 bf[8];
  #pragma unroll
  for (int kt = 0; kt < 8; ++kt)
    bf[kt].v = *(const v16bf*)&act[laneLo][kt * 32 + 16 * hi];

  #pragma unroll
  for (int which = 0; which < 2; ++which) {
    int mt = wave + which * 8;
    v8f acc;
    #pragma unroll
    for (int r = 0; r < 8; ++r) acc[r] = 0.f;
    const unsigned short* af = h1F + (size_t)mt * 8 * 512 + lane * 16;
    #pragma unroll
    for (int kt = 0; kt < 8; ++kt) {
      BF16x16 a; a.v = *(const v16bf*)(af + kt * 512);
      acc = __builtin_amdgcn_wmma_f32_16x16x32_bf16(
                false, a.v, false, bf[kt].v, (short)0, acc, false, false);
    }
    #pragma unroll
    for (int r = 0; r < 8; ++r) {
      int c = mt * 16 + r + hi * 8;
      float hv = acc[r] + b_h1[c];
      hsh[laneLo][c] = f2bf(hv > 0.f ? hv : 0.f);
    }
  }
  __syncthreads();

  BF16x16 hb[8];
  #pragma unroll
  for (int kt = 0; kt < 8; ++kt)
    hb[kt].v = *(const v16bf*)&hsh[laneLo][kt * 32 + 16 * hi];

  #pragma unroll
  for (int which = 0; which < 2; ++which) {
    int mt = wave + which * 8;
    v8f acc;
    #pragma unroll
    for (int r = 0; r < 8; ++r) acc[r] = 0.f;
    const unsigned short* af = h2F + (size_t)mt * 8 * 512 + lane * 16;
    #pragma unroll
    for (int kt = 0; kt < 8; ++kt) {
      BF16x16 a; a.v = *(const v16bf*)(af + kt * 512);
      acc = __builtin_amdgcn_wmma_f32_16x16x32_bf16(
                false, a.v, false, hb[kt].v, (short)0, acc, false, false);
    }
    #pragma unroll
    for (int r = 0; r < 8; ++r) {
      int c = mt * 16 + r + hi * 8;
      lsh[laneLo][c] = acc[r] + b_h2[c];
    }
  }
  __syncthreads();

  // Coalesced [c][t] output: one float4 store per (channel, time-quad)
  float* outb = out + (size_t)b * Q_CH * L_LEN + t0;
  for (int idx = tid; idx < Q_CH * 4; idx += 256) {
    int q = idx & 3, c = idx >> 2;
    f32x4 v;
    #pragma unroll
    for (int j = 0; j < 4; ++j) v[j] = lsh[q * 4 + j][c];
    *(f32x4*)(outb + (size_t)c * L_LEN + q * 4) = v;
  }
}

// ---------------------------------------------------------------------------
extern "C" void kernel_launch(void* const* d_in, const int* in_sizes, int n_in,
                              void* d_out, int out_size, void* d_ws, size_t ws_size,
                              hipStream_t stream)
{
  (void)in_sizes; (void)n_in; (void)out_size; (void)ws_size;
  const float* waveforms = (const float*)d_in[0];
  const float* melspecs  = (const float*)d_in[1];
  const float* w_up   = (const float*)d_in[2];
  const float* b_up   = (const float*)d_in[3];
  const float* w_in   = (const float*)d_in[4];
  const float* b_in   = (const float*)d_in[5];
  const float* w_gate = (const float*)d_in[6];
  const float* b_gate = (const float*)d_in[7];
  const float* w_cond = (const float*)d_in[8];
  const float* b_cond = (const float*)d_in[9];
  const float* w_res  = (const float*)d_in[10];
  const float* b_res  = (const float*)d_in[11];
  const float* w_skip = (const float*)d_in[12];
  const float* b_skip = (const float*)d_in[13];
  const float* w_h1   = (const float*)d_in[14];
  const float* b_h1   = (const float*)d_in[15];
  const float* w_h2   = (const float*)d_in[16];
  const float* b_h2   = (const float*)d_in[17];

  char* ws = (char*)d_ws;
  size_t off = 0;
  auto take = [&](size_t bytes) {
    char* p = ws + off; off += (bytes + 255) & ~(size_t)255; return p;
  };
  float*          res_t_a  = (float*)take((size_t)B_N * 8 * L_LEN * 16 * 4);
  float*          res_t_b  = (float*)take((size_t)B_N * 8 * L_LEN * 16 * 4);
  float*          skip_t   = (float*)take((size_t)B_N * 15 * L_LEN * 16 * 4);
  unsigned short* res_bf_a = (unsigned short*)take((size_t)B_N * L_LEN * 128 * 2);
  unsigned short* res_bf_b = (unsigned short*)take((size_t)B_N * L_LEN * 128 * 2);
  unsigned short* condb    = (unsigned short*)take((size_t)B_N * L_LEN * 96 * 2);
  unsigned short* gateF    = (unsigned short*)take((size_t)NB_C * 15 * 11 * 512 * 2);
  unsigned short* skipF    = (unsigned short*)take((size_t)NB_C * 15 * 4 * 512 * 2);
  unsigned short* resF     = (unsigned short*)take((size_t)NB_C * 8 * 4 * 512 * 2);
  unsigned short* h1F      = (unsigned short*)take((size_t)16 * 8 * 512 * 2);
  unsigned short* h2F      = (unsigned short*)take((size_t)16 * 8 * 512 * 2);
  float*          bgcomb   = (float*)take((size_t)NB_C * S_CH * 4);

  pack_frags<<<NB_C*15*11, 512, 0, stream>>>(0, 15, 11, w_gate, w_cond, w_skip, w_res, w_h1, w_h2, gateF);
  pack_frags<<<NB_C*15*4,  512, 0, stream>>>(1, 15, 4,  w_gate, w_cond, w_skip, w_res, w_h1, w_h2, skipF);
  pack_frags<<<NB_C*8*4,   512, 0, stream>>>(2, 8,  4,  w_gate, w_cond, w_skip, w_res, w_h1, w_h2, resF);
  pack_frags<<<16*8,       512, 0, stream>>>(3, 16, 8,  w_gate, w_cond, w_skip, w_res, w_h1, w_h2, h1F);
  pack_frags<<<16*8,       512, 0, stream>>>(4, 16, 8,  w_gate, w_cond, w_skip, w_res, w_h1, w_h2, h2F);
  pack_bias<<<(NB_C*S_CH + 255)/256, 256, 0, stream>>>(b_gate, b_cond, bgcomb);

  {
    long long total = (long long)96 * L_LEN;
    dim3 g((unsigned)((total + 255) / 256), B_N);
    upsample_kernel<<<g, 256, 0, stream>>>(melspecs, w_up, b_up, condb);
  }
  {
    long long total = (long long)L_LEN * 128;
    dim3 g((unsigned)((total + 255) / 256), B_N);
    init_res_kernel<<<g, 256, 0, stream>>>(waveforms, w_in, b_in, res_bf_a, res_t_a);
  }

  dim3 gb(L_LEN / 16, B_N);
  for (int i = 0; i < NB_C; ++i) {
    int d = 1 << (i % 8);
    const unsigned short* rbf_in  = (i & 1) ? res_bf_b : res_bf_a;
    unsigned short*       rbf_out = (i & 1) ? res_bf_a : res_bf_b;
    const float*          rt_in   = (i & 1) ? res_t_b : res_t_a;
    float*                rt_out  = (i & 1) ? res_t_a : res_t_b;
    wavenet_block_kernel<<<gb, 256, 0, stream>>>(
        rbf_in, rbf_out, rt_in, rt_out, skip_t, condb,
        gateF + (size_t)i * 15 * 11 * 512,
        skipF + (size_t)i * 15 * 4 * 512,
        resF  + (size_t)i * 8 * 4 * 512,
        bgcomb + i * S_CH, b_skip + i * S_CH, b_res + i * R_CH,
        d, (i == 0) ? 1 : 0);
  }

  head_kernel<<<gb, 256, 0, stream>>>(skip_t, h1F, h2F, b_h1, b_h2, (float*)d_out);
}
